// MoEHead_59631325938212
// MI455X (gfx1250) — matrix operand order repaired and checked by reading, not compiled, not run on hardware
//
#include <hip/hip_runtime.h>
#include <math.h>

// MoE head: B=16384 tokens, D=512, E=16 experts, H=2048, C=1 (top-1 routing).
constexpr int B = 16384;
constexpr int D = 512;
constexpr int E = 16;
constexpr int H = 2048;
constexpr int KP = D / 2;            // 256 K-pairs per column
constexpr float LN_EPS = 1e-5f;

typedef __attribute__((ext_vector_type(16))) __bf16 v16bf;
typedef __attribute__((ext_vector_type(8)))  float  v8f;

// fp32 -> bf16 round-to-nearest-even (bit trick), and back.
__device__ __forceinline__ unsigned short f2bf(float f) {
    unsigned int x = __float_as_uint(f);
    unsigned int r = x + 0x7FFFu + ((x >> 16) & 1u);
    return (unsigned short)(r >> 16);
}
__device__ __forceinline__ float bf2f(unsigned short h) {
    return __uint_as_float(((unsigned int)h) << 16);
}

struct Frag {
    union { v16bf v; unsigned short h[16]; unsigned int w[8]; };
};

__device__ __forceinline__ void load_bpk(const unsigned int* p, Frag& f) {
    *(uint4*)&f.w[0] = *(const uint4*)(p);
    *(uint4*)&f.w[4] = *(const uint4*)(p + 4);
}

__device__ __forceinline__ void wmma3(const Frag& ah, const Frag& al,
                                      const Frag& bh, const Frag& bl, v8f& c) {
    // bf16x3 split: ~fp32 accuracy, 3 WMMAs per 32-K chunk.
    c = __builtin_amdgcn_wmma_f32_16x16x32_bf16(false, ah.v, false, bh.v,
                                                (short)0, c, false, false);
    c = __builtin_amdgcn_wmma_f32_16x16x32_bf16(false, al.v, false, bh.v,
                                                (short)0, c, false, false);
    c = __builtin_amdgcn_wmma_f32_16x16x32_bf16(false, ah.v, false, bl.v,
                                                (short)0, c, false, false);
}

// ---------------------------------------------------------------------------
// Kernel 0: pack w1 (E,D,H) fp32 into bf16 hi/lo planes, layout
// [e][col][kpair] as u32 = (bf16(w1[2kp+1][col])<<16) | bf16(w1[2kp][col]).
// This is exactly the WMMA B-fragment register image -> GEMM kernel loads
// fragments with plain b128 loads, no conversion VALU in the hot loop.
// ---------------------------------------------------------------------------
__global__ __launch_bounds__(256)
void pack_w1_kernel(const float* __restrict__ w1,
                    unsigned int* __restrict__ hi_out,
                    unsigned int* __restrict__ lo_out)
{
    const int e  = blockIdx.z;
    const int c0 = blockIdx.y * 64;    // 64-column tile
    const int k0 = blockIdx.x * 128;   // 128-K-row tile (64 kpairs)

    __shared__ unsigned short shi[128][66];
    __shared__ unsigned short slo[128][66];

    const float* w1e = w1 + (size_t)e * D * H;
    const int tid = threadIdx.x;

#pragma unroll
    for (int it = 0; it < 32; ++it) {           // 128x64 coalesced reads
        const int idx = tid + it * 256;
        const int r = idx >> 6, cc = idx & 63;
        const float f = w1e[(size_t)(k0 + r) * H + c0 + cc];
        const unsigned short hh = f2bf(f);
        shi[r][cc] = hh;
        slo[r][cc] = f2bf(f - bf2f(hh));
    }
    __syncthreads();

#pragma unroll
    for (int it = 0; it < 16; ++it) {           // 64x64 contiguous writes
        const int idx = tid + it * 256;
        const int cc = idx >> 6, kp = idx & 63;
        const unsigned int hv =
            ((unsigned int)shi[2 * kp + 1][cc] << 16) | shi[2 * kp][cc];
        const unsigned int lv =
            ((unsigned int)slo[2 * kp + 1][cc] << 16) | slo[2 * kp][cc];
        const size_t o = ((size_t)e * H + c0 + cc) * KP + (k0 >> 1) + kp;
        hi_out[o] = hv;
        lo_out[o] = lv;
    }
}

// ---------------------------------------------------------------------------
// Kernel 1: fused gate GEMM + softmax + argmax routing + LayerNorm stats.
// One wave (32 lanes) per token; 8 waves per block. Memory bound on z.
// ---------------------------------------------------------------------------
__global__ __launch_bounds__(256)
void gate_ln_kernel(const float* __restrict__ z,
                    const float* __restrict__ gate_w,   // (D,E)
                    const float* __restrict__ gate_b,   // (E)
                    float* __restrict__ g_out,          // (B,E)
                    int*   __restrict__ cnt,            // (E)
                    int*   __restrict__ tok_list,       // (E,B)
                    float* __restrict__ mu_out,         // (B)
                    float* __restrict__ rs_out)         // (B)
{
    const int tid  = threadIdx.x;
    const int wv   = tid >> 5;
    const int lane = tid & 31;
    const int token = blockIdx.x * 8 + wv;

    const float4* zr = reinterpret_cast<const float4*>(z + (size_t)token * D);
    float4 zv[4];
    float s = 0.f, ss = 0.f;
#pragma unroll
    for (int i = 0; i < 4; ++i) {
        zv[i] = zr[lane + 32 * i];
        s  += zv[i].x + zv[i].y + zv[i].z + zv[i].w;
        ss += zv[i].x * zv[i].x + zv[i].y * zv[i].y
            + zv[i].z * zv[i].z + zv[i].w * zv[i].w;
    }
#pragma unroll
    for (int m = 1; m < 32; m <<= 1) { s += __shfl_xor(s, m, 32); ss += __shfl_xor(ss, m, 32); }
    const float mu   = s  * (1.f / D);
    const float var  = ss * (1.f / D) - mu * mu;
    const float rsig = rsqrtf(var + LN_EPS);

    float acc[E];
#pragma unroll
    for (int e = 0; e < E; ++e) acc[e] = 0.f;
#pragma unroll
    for (int i = 0; i < 4; ++i) {
        const int q  = lane + 32 * i;
        const int k0 = q * 4;
        const float zc[4] = { zv[i].x, zv[i].y, zv[i].z, zv[i].w };
#pragma unroll
        for (int j = 0; j < 4; ++j) {
            const float sv = zc[j];
            const float* gwr = gate_w + (size_t)(k0 + j) * E;
#pragma unroll
            for (int e = 0; e < E; ++e) acc[e] += sv * gwr[e];
        }
    }
#pragma unroll
    for (int e = 0; e < E; ++e) {
#pragma unroll
        for (int m = 1; m < 32; m <<= 1) acc[e] += __shfl_xor(acc[e], m, 32);
        acc[e] += gate_b[e];
    }

    if (lane == 0) {
        float mx = acc[0]; int best = 0;
#pragma unroll
        for (int e = 1; e < E; ++e) if (acc[e] > mx) { mx = acc[e]; best = e; }
        float p[E]; float sum = 0.f;
#pragma unroll
        for (int e = 0; e < E; ++e) { p[e] = expf(acc[e] - mx); sum += p[e]; }
        const float inv = 1.f / sum;
        float* gr = g_out + (size_t)token * E;
#pragma unroll
        for (int e = 0; e < E; ++e) gr[e] = p[e] * inv;

        const int pos = atomicAdd(&cnt[best], 1);
        tok_list[(size_t)best * B + pos] = token;
        mu_out[token] = mu;
        rs_out[token] = rsig;
    }
}

// ---------------------------------------------------------------------------
// Kernel 2: per-expert fused  gelu((zn*s+b) @ w1 + b1) @ w2 + b2  via
// bf16x3-split WMMA (fp32-accurate). Block = 16 gathered tokens, full H.
// PACKED: pre-packed bf16 B planes, rolled K loop with explicit cross-
// iteration double buffering (keeps VGPRs low -> high occupancy, and keeps
// next chunk's loads in flight during current chunk's WMMAs).
// ---------------------------------------------------------------------------
template <bool PACKED>
__global__ __launch_bounds__(256)
void moe_expert_kernel(const float* __restrict__ z,
                       const float* __restrict__ ln_scale,  // (E,D)
                       const float* __restrict__ ln_bias,   // (E,D)
                       const float* __restrict__ w1,        // (E,D,H)
                       const unsigned int* __restrict__ w1hipk,  // packed planes
                       const unsigned int* __restrict__ w1lopk,
                       const float* __restrict__ b1,        // (E,H)
                       const float* __restrict__ w2,        // (E,H)  (C==1)
                       const float* __restrict__ b2,        // (E)
                       const int*   __restrict__ cnt,
                       const int*   __restrict__ tok_list,
                       const float* __restrict__ muA,
                       const float* __restrict__ rsA,
                       float* __restrict__ logits)          // (B)
{
    const int e    = blockIdx.y;
    const int tile = blockIdx.x;
    const int c    = cnt[e];
    int v = c - tile * 16;
    if (v <= 0) return;            // uniform exit: EXEC stays all-1s below
    if (v > 16) v = 16;

    constexpr int AS = D + 8;      // padded ushort row stride (bank spread)
    __shared__ unsigned short sAhi[16 * AS];
    __shared__ unsigned short sAlo[16 * AS];
    __shared__ int   s_tok[16];
    __shared__ float s_mu[16], s_rs[16], s_out[16];

    const int tid = threadIdx.x;
    if (tid < 16) {
        const int tk = (tid < v) ? tok_list[(size_t)e * B + tile * 16 + tid] : -1;
        s_tok[tid] = tk;
        s_mu[tid]  = (tk >= 0) ? muA[tk] : 0.f;
        s_rs[tid]  = (tk >= 0) ? rsA[tk] : 0.f;
        s_out[tid] = 0.f;
    }
    __syncthreads();

    // Stage A = zn*ln_scale+ln_bias, pre-split into bf16 hi/lo planes in LDS.
    const float* lnse = ln_scale + (size_t)e * D;
    const float* lnbe = ln_bias  + (size_t)e * D;
#pragma unroll
    for (int it = 0; it < 32; ++it) {
        const int idx = tid + it * 256;
        const int m = idx >> 9, d = idx & (D - 1);
        const int tk = s_tok[m];
        float x = 0.f;
        if (tk >= 0) {
            const float zz = z[(size_t)tk * D + d];
            x = (zz - s_mu[m]) * s_rs[m] * lnse[d] + lnbe[d];
        }
        const unsigned short hi = f2bf(x);
        sAhi[m * AS + d] = hi;
        sAlo[m * AS + d] = f2bf(x - bf2f(hi));
    }
    __syncthreads();

    const float* w1e = w1 + (size_t)e * D * H;
    const float* b1e = b1 + (size_t)e * H;
    const float* w2e = w2 + (size_t)e * H;

    const int wv = tid >> 5, lane = tid & 31;
    const int sub = lane >> 4, lr = lane & 15;
    const int aro = lr * AS;

    // A-fragment loader (contiguous K layout: lanes 0-15 K0-15,
    // lanes 16-31 K16-31); 16B-aligned -> ds_load_b128 pairs.
    auto load_a = [&](int k32, Frag& ah, Frag& al) {
        const int ab0 = k32 + sub * 16;
        const int ab1 = k32 + sub * 16 + 8;
#pragma unroll
        for (int j = 0; j < 8; ++j) {
            ah.h[j]     = sAhi[aro + ab0 + j];
            ah.h[8 + j] = sAhi[aro + ab1 + j];
            al.h[j]     = sAlo[aro + ab0 + j];
            al.h[8 + j] = sAlo[aro + ab1 + j];
        }
    };

    float osum[8];
#pragma unroll
    for (int r = 0; r < 8; ++r) osum[r] = 0.f;

#pragma unroll 1
    for (int pass = 0; pass < 16; ++pass) {
        const int n0  = pass * 128 + wv * 16;
        const int col = n0 + lr;
        v8f cacc = {0.f, 0.f, 0.f, 0.f, 0.f, 0.f, 0.f, 0.f};

        if constexpr (PACKED) {
            const unsigned int* ph = w1hipk + ((size_t)e * H + col) * KP + sub * 8;
            const unsigned int* pl = w1lopk + ((size_t)e * H + col) * KP + sub * 8;

            // Rolled loop, step 2: both buffer sets live across the back-edge
            // so the allocator cannot merge them; chunk kc+1 loads stay in
            // flight while chunk kc's WMMAs execute (partial loadcnt waits).
            Frag ah, al, bh0, bl0, bh1, bl1;
            load_bpk(ph, bh0);
            load_bpk(pl, bl0);
#pragma unroll 1
            for (int kc = 0; kc < 16; kc += 2) {
                const int o1 = (kc + 1) * 16;            // u32 offset, next chunk
                const int o2 = ((kc + 2) & 15) * 16;     // wraps -> no OOB
                load_bpk(ph + o1, bh1);
                load_bpk(pl + o1, bl1);
                load_a(kc * 32, ah, al);
                wmma3(ah, al, bh0, bl0, cacc);
                load_bpk(ph + o2, bh0);
                load_bpk(pl + o2, bl0);
                load_a((kc + 1) * 32, ah, al);
                wmma3(ah, al, bh1, bl1, cacc);
            }
        } else {
#pragma unroll 1
            for (int kc = 0; kc < 16; ++kc) {
                const int k32 = kc * 32;
                Frag ah, al, bh, bl;
                load_a(k32, ah, al);
                const int rbase = k32 + sub * 16;
                if (k32 + 32 < D)
                    __builtin_prefetch(&w1e[(size_t)(k32 + 32 + sub * 16) * H + col], 0, 1);
#pragma unroll
                for (int j = 0; j < 16; ++j) {
                    const float f = w1e[(size_t)(rbase + j) * H + col];
                    const unsigned short hh = f2bf(f);
                    bh.h[j] = hh;
                    bl.h[j] = f2bf(f - bf2f(hh));
                }
                wmma3(ah, al, bh, bl, cacc);
            }
        }

        // Fused epilogue: +b1, exact GELU, * w2 column, accumulate over H.
        const float bb = b1e[col];
        const float wc = w2e[col];
#pragma unroll
        for (int r = 0; r < 8; ++r) {
            const float hval = cacc[r] + bb;
            const float gl = 0.5f * hval * (1.f + erff(hval * 0.70710678118654752f));
            osum[r] += gl * wc;
        }
    }

    // Reduce over the 16 N-lanes of each half-wave, then across 8 waves.
#pragma unroll
    for (int r = 0; r < 8; ++r) {
#pragma unroll
        for (int m = 1; m < 16; m <<= 1) osum[r] += __shfl_xor(osum[r], m, 32);
    }
    if (lr == 0) {
#pragma unroll
        for (int r = 0; r < 8; ++r) atomicAdd(&s_out[r + sub * 8], osum[r]);
    }
    __syncthreads();

    if (tid < 16 && s_tok[tid] >= 0)
        logits[s_tok[tid]] = s_out[tid] + b2[e];
}

// ---------------------------------------------------------------------------
extern "C" void kernel_launch(void* const* d_in, const int* in_sizes, int n_in,
                              void* d_out, int out_size, void* d_ws, size_t ws_size,
                              hipStream_t stream) {
    const float* z        = (const float*)d_in[0];
    const float* gate_w   = (const float*)d_in[1];
    const float* gate_b   = (const float*)d_in[2];
    const float* ln_scale = (const float*)d_in[3];
    const float* ln_bias  = (const float*)d_in[4];
    const float* w1       = (const float*)d_in[5];
    const float* b1       = (const float*)d_in[6];
    const float* w2       = (const float*)d_in[7];
    const float* b2       = (const float*)d_in[8];

    float* logits = (float*)d_out;        // (B,)
    float* g_out  = logits + B;           // (B,E)

    // Workspace carve-up.
    char* ws = (char*)d_ws;
    int*   cnt      = (int*)ws;                               // E ints (64B slot)
    int*   tok_list = (int*)(ws + 64);                        // E*B ints
    float* mu_arr   = (float*)(ws + 64 + (size_t)E * B * 4);  // B floats
    float* rs_arr   = mu_arr + B;                             // B floats
    const size_t route_bytes = 64 + (size_t)E * B * 4 + 2 * (size_t)B * 4;
    const size_t plane_elems = (size_t)E * H * KP;            // u32 each
    unsigned int* w1hipk = (unsigned int*)(ws + route_bytes);
    unsigned int* w1lopk = w1hipk + plane_elems;
    const size_t req = route_bytes + 2 * plane_elems * 4;     // ~68.5 MB
    const bool packed = (ws_size >= req);

    hipMemsetAsync(cnt, 0, 64, stream);

    if (packed) {
        dim3 gp(D / 128, H / 64, E);      // (4, 32, 16)
        pack_w1_kernel<<<gp, 256, 0, stream>>>(w1, w1hipk, w1lopk);
    }

    gate_ln_kernel<<<B / 8, 256, 0, stream>>>(z, gate_w, gate_b, g_out,
                                              cnt, tok_list, mu_arr, rs_arr);

    dim3 grid2(B / 16, E);  // worst case: all tokens on one expert
    if (packed) {
        moe_expert_kernel<true><<<grid2, 256, 0, stream>>>(
            z, ln_scale, ln_bias, w1, w1hipk, w1lopk, b1, w2, b2,
            cnt, tok_list, mu_arr, rs_arr, logits);
    } else {
        moe_expert_kernel<false><<<grid2, 256, 0, stream>>>(
            z, ln_scale, ln_bias, w1, w1hipk, w1lopk, b1, w2, b2,
            cnt, tok_list, mu_arr, rs_arr, logits);
    }
}
